// TinyViTBlock_61289183314178
// MI455X (gfx1250) — compile-verified, matrix operand order checked by reading, not run on hardware
//
#include <hip/hip_runtime.h>
#include <hip/hip_bf16.h>

// ---------------------------------------------------------------------------
// TinyViT block for MI455X (gfx1250): all GEMMs + attention on
// v_wmma_f32_16x16x32_f16 (wave32 WMMA), f16 activations, f32 accumulation.
// GEMM A tiles stream via global_load_async_to_lds_b128 (ASYNCcnt) with a
// double-buffered LDS pipeline.
// ---------------------------------------------------------------------------

typedef __attribute__((ext_vector_type(16))) _Float16 v16h;
typedef __attribute__((ext_vector_type(8)))  _Float16 v8h;
typedef __attribute__((ext_vector_type(8)))  float    v8f;

#define BDIM 384          // C
#define HID  1536
#define NTOK 100352       // B*N = 32*3136
#define NWIN 2048         // 32 * 8 * 8
#define NH_  6

// --- index maps between (b,h,w) order and window-partition order -----------
__device__ __forceinline__ int winrev(int rw) {     // window order -> x order
    int b   = rw / 3136;
    int rem = rw - b * 3136;
    int w   = rem / 49;
    int t   = rem - w * 49;
    int hi = w >> 3, wj = w & 7;
    int ti = t / 7,  tj = t - ti * 7;
    return b * 3136 + (hi * 7 + ti) * 56 + (wj * 7 + tj);
}
__device__ __forceinline__ int winfwd(int rx) {     // x order -> window order
    int b = rx / 3136;
    int p = rx - b * 3136;
    int h = p / 56, w = p - h * 56;
    int hi = h / 7, ti = h - hi * 7;
    int wj = w / 7, tj = w - wj * 7;
    return ((b * 8 + hi) * 8 + wj) * 49 + ti * 7 + tj;
}

// --- WMMA 16x16x32 f16 fragment load from LDS ------------------------------
// A-frag layout (ISA 7.12.2): lane<16 holds row M=lane, elems 0-7 -> K 0..7,
// elems 8-15 -> K 16..23; lanes>=16: +8 on K. B-frag mirrors with N on lanes,
// so both load from [row][k] tiles with contiguous 16B chunks.
__device__ __forceinline__ v16h ldfrag(const _Float16* base, int pitch,
                                       int row0, int k0, int lane) {
    const _Float16* p = base + (size_t)(row0 + (lane & 15)) * pitch
                             + k0 + ((lane >> 4) << 3);
    v8h lo = *(const v8h*)(p);
    v8h hi = *(const v8h*)(p + 16);
    return __builtin_shufflevector(lo, hi, 0,1,2,3,4,5,6,7,
                                          8,9,10,11,12,13,14,15);
}

// --- CDNA5 async copy global->LDS (16B per lane, ASYNCcnt tracked) ---------
__device__ __forceinline__ void async_cp16(const _Float16* gsrc,
                                           _Float16* ldst) {
    unsigned la = (unsigned)(unsigned long long)(size_t)ldst;   // addr[31:0] = LDS offset
    unsigned long long ga = (unsigned long long)(size_t)gsrc;
    asm volatile("global_load_async_to_lds_b128 %0, %1, off"
                 :: "v"(la), "v"(ga) : "memory");
}
__device__ __forceinline__ void async_wait2() {
    asm volatile("s_wait_asynccnt 0x2" ::: "memory");
}
__device__ __forceinline__ void async_wait0() {
    asm volatile("s_wait_asynccnt 0x0" ::: "memory");
}

// --- fp32 -> f16 weight convert --------------------------------------------
__global__ __launch_bounds__(256)
void cvt_f16_kernel(const float* __restrict__ in, _Float16* __restrict__ out, int n) {
    int i = blockIdx.x * 256 + threadIdx.x;
    if (i < n) out[i] = (_Float16)in[i];
}

// --- LayerNorm over C=384, one wave per token ------------------------------
template<bool REORDER>
__global__ __launch_bounds__(256)
void ln_kernel(const float* __restrict__ x, const float* __restrict__ g,
               const float* __restrict__ be, _Float16* __restrict__ out) {
    const int wid  = threadIdx.x >> 5;
    const int lane = threadIdx.x & 31;
    const int token = blockIdx.x * 8 + wid;                 // < NTOK exactly
    const float* xr = x + (size_t)token * BDIM;
    float v[12], s = 0.f, s2 = 0.f;
#pragma unroll
    for (int i = 0; i < 12; ++i) {
        v[i] = xr[lane + 32 * i];
        s += v[i]; s2 += v[i] * v[i];
    }
#pragma unroll
    for (int m = 16; m; m >>= 1) {
        s  += __shfl_xor(s,  m, 32);
        s2 += __shfl_xor(s2, m, 32);
    }
    const float mu   = s * (1.f / BDIM);
    const float var  = s2 * (1.f / BDIM) - mu * mu;
    const float rstd = rsqrtf(var + 1e-5f);
    const int orow = REORDER ? winfwd(token) : token;
    _Float16* o = out + (size_t)orow * BDIM;
#pragma unroll
    for (int i = 0; i < 12; ++i) {
        int c = lane + 32 * i;
        o[c] = (_Float16)((v[i] - mu) * rstd * g[c] + be[c]);
    }
}

// --- tiled WMMA GEMM: C[M,N] = A[M,K](f16) * B[K,N](f16), templated epilogue
// EPI 0: out f16 = acc                      (qkv)
// EPI 1: out f32 = acc + bias               (fc1)
// EPI 2: out f32[winrev(row)] = acc + bias + res[winrev(row)]   (proj+resid)
// EPI 3: out f32 = acc + bias + res[row]    (fc2+resid)
#define BM 128
#define BN 128
#define BK 32
#define GT 256
template<int EPI>
__global__ __launch_bounds__(GT)
void gemm_wmma(const _Float16* __restrict__ A, const _Float16* __restrict__ B,
               const float* __restrict__ bias, const float* __restrict__ res,
               float* __restrict__ outF, _Float16* __restrict__ outH,
               int N, int K) {
    __shared__ alignas(16) _Float16 As[2][BM][BK + 8];
    __shared__ alignas(16) _Float16 Bs[2][BN][BK + 8];
    const int t    = threadIdx.x;
    const int lane = t & 31;
    const int wid  = t >> 5;
    const int wm   = wid & 3;         // 4 waves over M (32 rows each)
    const int wn   = wid >> 2;        // 2 waves over N (64 cols each)
    const int bm   = blockIdx.y * BM;
    const int bn   = blockIdx.x * BN;

    v8f acc[2][4];
#pragma unroll
    for (int i = 0; i < 2; ++i)
#pragma unroll
        for (int j = 0; j < 4; ++j) acc[i][j] = {};

    // stage A: 512 async 16B chunks (2 per thread); r = chunk>>2, kcol 0/8/16/24
    auto stageA = [&](int buf, int k0) {
#pragma unroll
        for (int i = 0; i < 2; ++i) {
            int lin = t + GT * i;                 // 0..511
            int r  = lin >> 2;
            int kc = (lin & 3) << 3;
            async_cp16(A + (size_t)(bm + r) * K + k0 + kc, &As[buf][r][kc]);
        }
    };
    // stage B transposed -> Bs[n][k] via VGPRs (N-contiguous global reads)
    auto stageB = [&](int buf, int k0) {
#pragma unroll
        for (int i = 0; i < 16; ++i) {
            int lin = t + GT * i;
            int k = lin >> 7, n = lin & 127;
            Bs[buf][n][k] = B[(size_t)(k0 + k) * N + bn + n];
        }
        if (k0 + BK < K)   // global_prefetch_b8 of next B tile
            __builtin_prefetch(B + (size_t)(k0 + BK) * N + bn + (t & 127), 0, 1);
    };

    const int KT = K / BK;
    stageA(0, 0);
    stageB(0, 0);
    for (int kt = 0; kt < KT; ++kt) {
        const int cur = kt & 1;
        if (kt + 1 < KT) {                 // fill the other buffer
            stageA(cur ^ 1, (kt + 1) * BK);
            stageB(cur ^ 1, (kt + 1) * BK);
            async_wait2();                 // in-order: current stage resident
        } else {
            async_wait0();
        }
        __syncthreads();

        v16h af[2], bf[4];
#pragma unroll
        for (int mt = 0; mt < 2; ++mt)
            af[mt] = ldfrag(&As[cur][0][0], BK + 8, wm * 32 + mt * 16, 0, lane);
#pragma unroll
        for (int nt = 0; nt < 4; ++nt)
            bf[nt] = ldfrag(&Bs[cur][0][0], BK + 8, wn * 64 + nt * 16, 0, lane);
#pragma unroll
        for (int mt = 0; mt < 2; ++mt)
#pragma unroll
            for (int nt = 0; nt < 4; ++nt)
                acc[mt][nt] = __builtin_amdgcn_wmma_f32_16x16x32_f16(
                    false, af[mt], false, bf[nt], (short)0, acc[mt][nt],
                    false, false);
        __syncthreads();                   // all reads of `cur` done before refill
    }

    // epilogue: lane l holds N=l%16, rows r + 8*(l>=16)  (C/D layout)
#pragma unroll
    for (int mt = 0; mt < 2; ++mt)
#pragma unroll
        for (int nt = 0; nt < 4; ++nt) {
            const int col   = bn + wn * 64 + nt * 16 + (lane & 15);
            const int rbase = bm + wm * 32 + mt * 16 + ((lane >> 4) << 3);
            const float bv  = (EPI == 0) ? 0.f : bias[col];
#pragma unroll
            for (int r = 0; r < 8; ++r) {
                const int row = rbase + r;
                const float v = acc[mt][nt][r];
                if (EPI == 0) {
                    outH[(size_t)row * N + col] = (_Float16)v;
                } else if (EPI == 1) {
                    outF[(size_t)row * N + col] = v + bv;
                } else if (EPI == 2) {
                    const int rx = winrev(row);
                    outF[(size_t)rx * N + col] =
                        v + bv + res[(size_t)rx * N + col];
                } else {
                    outF[(size_t)row * N + col] =
                        v + bv + res[(size_t)row * N + col];
                }
            }
        }
}

// --- window attention: one wave per (window, head) -------------------------
__global__ __launch_bounds__(32)
void attn_kernel(const _Float16* __restrict__ qkv,
                 const float* __restrict__ relb,
                 _Float16* __restrict__ out) {
    __shared__ alignas(16) _Float16 Qs[64][72];   // Q, later reused as P
    __shared__ alignas(16) _Float16 Ks[64][72];
    __shared__ alignas(16) _Float16 Vt[64][72];   // V transposed: Vt[d][m]
    __shared__ float Ss[64][65];
    const int lane = threadIdx.x;
    const int head = blockIdx.x;            // 0..5
    const int win  = blockIdx.y;            // 0..2047
    const size_t qbase = (size_t)win * 49 * 1152 + head * 64;

    // stage Q/K/V, zero-pad rows 49..63
#pragma unroll 4
    for (int i = 0; i < 128; ++i) {
        int lin = lane + 32 * i;
        int n = lin >> 6, d = lin & 63;
        _Float16 q = (_Float16)0.f, k = (_Float16)0.f, vv = (_Float16)0.f;
        if (n < 49) {
            size_t o = qbase + (size_t)n * 1152 + d;
            q = qkv[o]; k = qkv[o + 384]; vv = qkv[o + 768];
        }
        Qs[n][d] = q; Ks[n][d] = k; Vt[d][n] = vv;
    }
    __syncthreads();

    // S = (Q * K^T) * scale  (K^T frag == contiguous rows of Ks)
#pragma unroll
    for (int mt = 0; mt < 4; ++mt)
#pragma unroll
        for (int nt = 0; nt < 4; ++nt) {
            v8f s = {};
#pragma unroll
            for (int kk = 0; kk < 64; kk += 32) {
                v16h a = ldfrag(&Qs[0][0], 72, mt * 16, kk, lane);
                v16h b = ldfrag(&Ks[0][0], 72, nt * 16, kk, lane);
                s = __builtin_amdgcn_wmma_f32_16x16x32_f16(
                        false, a, false, b, (short)0, s, false, false);
            }
            const int m  = nt * 16 + (lane & 15);
            const int nb = mt * 16 + ((lane >> 4) << 3);
#pragma unroll
            for (int r = 0; r < 8; ++r) Ss[nb + r][m] = s[r] * 0.125f;
        }
    __syncthreads();

    // softmax rows (+analytic rel-pos bias); write P over Qs
    for (int rep = 0; rep < 2; ++rep) {
        const int n = lane + 32 * rep;
        if (n < 49) {
            const int ni = n / 7, nj = n - (n / 7) * 7;
            float mx = -1e30f;
            for (int m = 0; m < 49; ++m) {
                int mi = m / 7, mj = m - mi * 7;
                int idx = (ni - mi + 6) * 13 + (nj - mj + 6);
                mx = fmaxf(mx, Ss[n][m] + relb[idx * NH_ + head]);
            }
            float sum = 0.f;
            for (int m = 0; m < 49; ++m) {
                int mi = m / 7, mj = m - mi * 7;
                int idx = (ni - mi + 6) * 13 + (nj - mj + 6);
                sum += __expf(Ss[n][m] + relb[idx * NH_ + head] - mx);
            }
            const float inv = 1.f / sum;
            for (int m = 0; m < 64; ++m) {
                float p = 0.f;
                if (m < 49) {
                    int mi = m / 7, mj = m - mi * 7;
                    int idx = (ni - mi + 6) * 13 + (nj - mj + 6);
                    p = __expf(Ss[n][m] + relb[idx * NH_ + head] - mx) * inv;
                }
                Qs[n][m] = (_Float16)p;
            }
        } else {
            for (int m = 0; m < 64; ++m) Qs[n][m] = (_Float16)0.f;
        }
    }
    __syncthreads();

    // O = P * V   (B frag columns d -> contiguous rows of Vt)
    const size_t obase = (size_t)win * 49 * 384 + head * 64;
#pragma unroll
    for (int mt = 0; mt < 4; ++mt)
#pragma unroll
        for (int nt = 0; nt < 4; ++nt) {
            v8f o = {};
#pragma unroll
            for (int kk = 0; kk < 64; kk += 32) {
                v16h a = ldfrag(&Qs[0][0], 72, mt * 16, kk, lane);
                v16h b = ldfrag(&Vt[0][0], 72, nt * 16, kk, lane);
                o = __builtin_amdgcn_wmma_f32_16x16x32_f16(
                        false, a, false, b, (short)0, o, false, false);
            }
            const int d  = nt * 16 + (lane & 15);
            const int nb = mt * 16 + ((lane >> 4) << 3);
#pragma unroll
            for (int r = 0; r < 8; ++r) {
                const int n = nb + r;
                if (n < 49)
                    out[obase + (size_t)n * 384 + d] = (_Float16)o[r];
            }
        }
}

// --- 3x3 depthwise conv + exact GELU ---------------------------------------
__global__ __launch_bounds__(256)
void dwgelu_kernel(const float* __restrict__ y1, const float* __restrict__ wdw,
                   const float* __restrict__ bdw, _Float16* __restrict__ y2) {
    const size_t tid = (size_t)blockIdx.x * 256 + threadIdx.x;
    const int ch  = (int)(tid % HID);
    const size_t row = tid / HID;
    const int b = (int)(row / 3136);
    const int p = (int)(row - (size_t)b * 3136);
    const int h = p / 56, w = p - (p / 56) * 56;
    float acc = bdw[ch];
    const float* wk = wdw + ch * 9;
#pragma unroll
    for (int dh = -1; dh <= 1; ++dh) {
        const int hh = h + dh;
        if (hh < 0 || hh >= 56) continue;
#pragma unroll
        for (int dw = -1; dw <= 1; ++dw) {
            const int ww = w + dw;
            if (ww < 0 || ww >= 56) continue;
            const size_t r2 = (size_t)b * 3136 + hh * 56 + ww;
            acc += y1[r2 * HID + ch] * wk[(dh + 1) * 3 + (dw + 1)];
        }
    }
    const float g = 0.5f * acc * (1.f + erff(acc * 0.70710678118f));
    y2[tid] = (_Float16)g;
}

// ---------------------------------------------------------------------------
extern "C" void kernel_launch(void* const* d_in, const int* in_sizes, int n_in,
                              void* d_out, int out_size, void* d_ws, size_t ws_size,
                              hipStream_t stream) {
    (void)in_sizes; (void)n_in; (void)out_size; (void)ws_size;
    const float* x      = (const float*)d_in[0];
    const float* gamma1 = (const float*)d_in[1];
    const float* beta1  = (const float*)d_in[2];
    const float* w_qkv  = (const float*)d_in[3];
    const float* w_proj = (const float*)d_in[4];
    const float* b_proj = (const float*)d_in[5];
    const float* relb   = (const float*)d_in[6];
    const float* gamma2 = (const float*)d_in[7];
    const float* beta2  = (const float*)d_in[8];
    const float* w_fc1  = (const float*)d_in[9];
    const float* b_fc1  = (const float*)d_in[10];
    const float* w_dw   = (const float*)d_in[11];
    const float* b_dw   = (const float*)d_in[12];
    const float* w_fc2  = (const float*)d_in[13];
    const float* b_fc2  = (const float*)d_in[14];
    float* out = (float*)d_out;

    // workspace layout (sequential-lifetime reuse)
    char* ws = (char*)d_ws;
    float*    x1   = (float*)   (ws + 0);                       // 154,140,672 B
    _Float16* bufA = (_Float16*)(ws + 154140672ULL);            //  77,070,336 B (xn/attnout/ln2)
    _Float16* bufB = (_Float16*)(ws + 231211008ULL);            // 308,281,344 B (qkv/y2)
    float*    y1   = (float*)   (ws + 539492352ULL);            // 616,562,688 B
    _Float16* wq16 = (_Float16*)(ws + 1156055040ULL);           //     884,736 B
    _Float16* wp16 = (_Float16*)(ws + 1156939776ULL);           //     294,912 B
    _Float16* wf1  = (_Float16*)(ws + 1157234688ULL);           //   1,179,648 B
    _Float16* wf2  = (_Float16*)(ws + 1158414336ULL);           //   1,179,648 B

    // 1) weights -> f16
    cvt_f16_kernel<<<(442368 + 255) / 256, 256, 0, stream>>>(w_qkv,  wq16, 442368);
    cvt_f16_kernel<<<(147456 + 255) / 256, 256, 0, stream>>>(w_proj, wp16, 147456);
    cvt_f16_kernel<<<(589824 + 255) / 256, 256, 0, stream>>>(w_fc1,  wf1,  589824);
    cvt_f16_kernel<<<(589824 + 255) / 256, 256, 0, stream>>>(w_fc2,  wf2,  589824);

    // 2) LN1 -> f16, window-partitioned order
    ln_kernel<true><<<NTOK / 8, 256, 0, stream>>>(x, gamma1, beta1, bufA);

    // 3) qkv = xn @ w_qkv : M=100352, N=1152, K=384
    gemm_wmma<0><<<dim3(1152 / BN, NTOK / BM), GT, 0, stream>>>(
        bufA, wq16, nullptr, nullptr, nullptr, bufB, 1152, 384);

    // 4) window attention: one wave per (window, head)
    attn_kernel<<<dim3(NH_, NWIN), 32, 0, stream>>>(bufB, relb, bufA);

    // 5) x1 = x + window_reverse(attnout @ w_proj + b_proj) : N=K=384
    gemm_wmma<2><<<dim3(384 / BN, NTOK / BM), GT, 0, stream>>>(
        bufA, wp16, b_proj, x, x1, nullptr, 384, 384);

    // 6) LN2 -> f16
    ln_kernel<false><<<NTOK / 8, 256, 0, stream>>>(x1, gamma2, beta2, bufA);

    // 7) y1 = ln2 @ w_fc1 + b_fc1 : N=1536, K=384
    gemm_wmma<1><<<dim3(1536 / BN, NTOK / BM), GT, 0, stream>>>(
        bufA, wf1, b_fc1, nullptr, y1, nullptr, 1536, 384);

    // 8) depthwise 3x3 + GELU -> f16
    dwgelu_kernel<<<(unsigned)((size_t)NTOK * HID / 256), 256, 0, stream>>>(
        y1, w_dw, b_dw, bufB);

    // 9) out = x1 + y2 @ w_fc2 + b_fc2 : N=384, K=1536
    gemm_wmma<3><<<dim3(384 / BN, NTOK / BM), GT, 0, stream>>>(
        bufB, wf2, b_fc2, x1, out, nullptr, 384, 1536);
}
